// EEGTransformer_27608049779322
// MI455X (gfx1250) — matrix operand order repaired and checked by reading, compiled, not run on hardware
//
#include <hip/hip_runtime.h>
#include <hip/hip_bf16.h>

typedef _Float16 h16;
typedef __attribute__((ext_vector_type(16))) _Float16 v16h;
typedef __attribute__((ext_vector_type(8)))  _Float16 v8h;
typedef __attribute__((ext_vector_type(4)))  _Float16 v4h;
typedef __attribute__((ext_vector_type(8)))  float    v8f;
typedef __attribute__((ext_vector_type(4)))  unsigned int tdm_v4u;
typedef __attribute__((ext_vector_type(8)))  int          tdm_v8i;
typedef __attribute__((ext_vector_type(4)))  int          tdm_v4i;

#define NE  128          // n_embd
#define NT  512          // block_size
#define NB  64           // batch
#define NH  4            // heads
#define NHS 32           // head_size
#define NL  4            // layers
#define NV  512          // vocab
#define NM  (NB*NT)      // 32768 tokens

// ---------------------------------------------------------------- fragments
// 16-bit A fragment (16x32, row-major source, ld in halfs):
// lane = {hi,m}: halves 0..15 -> K = kb+16*hi+0..15 (contiguous 32B)
static __device__ __forceinline__ v16h ld_a16(const h16* p) {
  return *(const v16h*)p;
}
// 16-bit B fragment (32x16) from Bt[N,K] row-major: lane column n=lane&15,
// halves 0..15 -> K = kb + 16*(lane>>4) + 0..15 (contiguous 32B)
static __device__ __forceinline__ v16h ld_b16(const h16* p) {
  return *(const v16h*)p;
}
static __device__ __forceinline__ v8f wmma32(v16h a, v16h b, v8f c) {
  return __builtin_amdgcn_wmma_f32_16x16x32_f16(false, a, false, b, (short)0, c,
                                                false, false);
}

// ---------------------------------------------------------------- prep
__global__ __launch_bounds__(256) void pack_qkv_f16(
    const float* __restrict__ wq, const float* __restrict__ wk,
    const float* __restrict__ wv, h16* __restrict__ dst) {
  // dst: [L][384][E] f16; rows 0..127 = Q (h*32+d), 128..255 = K, 256..383 = V
  int id = blockIdx.x * 256 + threadIdx.x;
  if (id >= NL * 384 * NE) return;
  int e = id & (NE - 1);
  int r = (id >> 7) % 384;
  int l = id / (384 * NE);
  int sect = r >> 7;
  int rr = r & 127;
  int h = rr >> 5, d = rr & 31;
  const float* src = (sect == 0) ? wq : (sect == 1) ? wk : wv;
  float v = src[(((size_t)(l * NH + h) * NE) + e) * NHS + d];
  dst[id] = (h16)v;
}

__global__ __launch_bounds__(256) void cast_f16(const float* __restrict__ s,
                                                h16* __restrict__ d, int n) {
  for (int i = blockIdx.x * 256 + threadIdx.x; i < n; i += gridDim.x * 256)
    d[i] = (h16)s[i];
}

// ---------------------------------------------------------------- embedding
__global__ __launch_bounds__(256) void embed_k(const int* __restrict__ idx,
                                               const float* __restrict__ tok_emb,
                                               const float* __restrict__ pos_emb,
                                               float* __restrict__ x) {
  int id = blockIdx.x * 256 + threadIdx.x;        // over NM * (NE/4)
  int bt = id >> 5;
  int j = id & 31;
  int t = bt & (NT - 1);
  int token = idx[bt];
  float4 a = ((const float4*)(tok_emb + (size_t)token * NE))[j];
  float4 p = ((const float4*)(pos_emb + (size_t)t * NE))[j];
  float4 o; o.x = a.x + p.x; o.y = a.y + p.y; o.z = a.z + p.z; o.w = a.w + p.w;
  ((float4*)(x + (size_t)bt * NE))[j] = o;
}

// ---------------------------------------------------------------- layernorm
__global__ __launch_bounds__(256) void ln_f16(const float* __restrict__ x,
                                              const float* __restrict__ g,
                                              const float* __restrict__ b,
                                              h16* __restrict__ o) {
  int wave = threadIdx.x >> 5, lane = threadIdx.x & 31;
  int tok = blockIdx.x * 8 + wave;
  float4 v = ((const float4*)(x + (size_t)tok * NE))[lane];
  float s = v.x + v.y + v.z + v.w;
#pragma unroll
  for (int d = 16; d >= 1; d >>= 1) s += __shfl_xor(s, d, 32);
  float mean = s * (1.0f / NE);
  float d0 = v.x - mean, d1 = v.y - mean, d2 = v.z - mean, d3 = v.w - mean;
  float q = d0 * d0 + d1 * d1 + d2 * d2 + d3 * d3;
#pragma unroll
  for (int d = 16; d >= 1; d >>= 1) q += __shfl_xor(q, d, 32);
  float rstd = rsqrtf(q * (1.0f / NE) + 1e-5f);
  int c = lane * 4;
  v4h ov;
  ov[0] = (h16)(d0 * rstd * g[c + 0] + b[c + 0]);
  ov[1] = (h16)(d1 * rstd * g[c + 1] + b[c + 1]);
  ov[2] = (h16)(d2 * rstd * g[c + 2] + b[c + 2]);
  ov[3] = (h16)(d3 * rstd * g[c + 3] + b[c + 3]);
  *(v4h*)(o + (size_t)tok * NE + c) = ov;
}

// ---------------------------------------------------------------- GEMM
// C[M,N] = A[M,K] * Bt[N,K]^T ; 8 waves/block, wave = one 16x32 slab
// (two 16x16 tiles sharing the A fragment). Block = 128 rows x 32 cols,
// grid = (M/128, N/32). The 32-wide B tile (shared by all 8 waves) is
// DMA'd into LDS by the Tensor Data Mover (tensor_load_to_lds, ISA §8).
enum { GM_QKV = 0, GM_RESID = 1, GM_GELU = 2, GM_OUT = 3 };

template <int MODE>
__global__ __launch_bounds__(256) void gemm_ws(
    const h16* __restrict__ A, const h16* __restrict__ Bt,
    const float* __restrict__ bias, float* __restrict__ outF,
    h16* __restrict__ outH, h16* __restrict__ qo, h16* __restrict__ ko,
    h16* __restrict__ vto, int Kd) {
  __shared__ __align__(32) h16 bbuf[32 * 512];   // B tile [32][Kd]
  int lane = threadIdx.x & 31;
  int wave = threadIdx.x >> 5;
  int m = lane & 15;
  int hb = lane >> 4;
  int row0 = blockIdx.x * 128 + wave * 16;
  int col0 = blockIdx.y * 32;

  if (threadIdx.x < 32) {
    // Tensor DMA descriptor: 2D tile [y=32 rows][x=Kd elems], 2B elems.
    unsigned long long ga =
        (unsigned long long)(uintptr_t)(Bt + (size_t)col0 * Kd);
    unsigned int ldsoff = (unsigned int)(uintptr_t)(void*)bbuf; // LDS byte addr
    tdm_v4u g0;
    g0[0] = 1u;                                   // count=1, user mode
    g0[1] = ldsoff;                               // lds_addr
    g0[2] = (unsigned int)ga;                     // global_addr[31:0]
    g0[3] = (unsigned int)((ga >> 32) & 0x01FFFFFFu) | (2u << 30); // type=2
    tdm_v8i g1;
    g1[0] = 0x00010000;                           // data_size=1 (2 bytes)
    g1[1] = (Kd & 0xffff) << 16;                  // tensor_dim0[15:0]
    g1[2] = ((Kd >> 16) & 0xffff) | (32 << 16);   // dim0 hi | tensor_dim1=32
    g1[3] = (Kd << 16);                           // dim1 hi=0 | tile_dim0=Kd
    g1[4] = 32;                                   // tile_dim1=32, tile_dim2=0
    g1[5] = Kd;                                   // tensor_dim0_stride lo
    g1[6] = 0;                                    // stride hi | dim1_stride lo
    g1[7] = 0;
    tdm_v4i g2 = {0, 0, 0, 0};
    tdm_v4i g3 = {0, 0, 0, 0};
    tdm_v8i g4 = {0, 0, 0, 0, 0, 0, 0, 0};        // 6-arg toolchain variant
    __builtin_amdgcn_tensor_load_to_lds(g0, g1, g2, g3, g4, 0);
    __builtin_amdgcn_s_wait_tensorcnt(0);
  }
  __syncthreads();

  const h16* ap  = A + (size_t)(row0 + m) * Kd + 16 * hb;
  const h16* bp0 = bbuf + (size_t)m * Kd + 16 * hb;         // cols col0+0..15
  const h16* bp1 = bbuf + (size_t)(16 + m) * Kd + 16 * hb;  // cols col0+16..31
  v8f acc0 = {}, acc1 = {};
#pragma unroll 4
  for (int kb = 0; kb < Kd; kb += 32) {
    v16h af = ld_a16(ap + kb);
    acc0 = wmma32(af, ld_b16(bp0 + kb), acc0);
    acc1 = wmma32(af, ld_b16(bp1 + kb), acc1);
  }

#pragma unroll
  for (int n = 0; n < 2; ++n) {
    int col = col0 + 16 * n + m;
#pragma unroll
    for (int i = 0; i < 8; ++i) {
      int row = row0 + hb * 8 + i;
      float v = (n == 0) ? acc0[i] : acc1[i];
      if constexpr (MODE == GM_QKV) {
        int sect = col >> 7, cc = col & 127;
        int hh = cc >> 5, dd = cc & 31;
        int b = row >> 9, t = row & (NT - 1);
        int bh = b * NH + hh;
        h16 hv = (h16)v;
        if (sect == 0)      qo[((size_t)bh * NT + t) * NHS + dd] = hv;
        else if (sect == 1) ko[((size_t)bh * NT + t) * NHS + dd] = hv;
        else                vto[((size_t)bh * NHS + dd) * NT + t] = hv;  // V^T
      } else if constexpr (MODE == GM_RESID) {
        outF[(size_t)row * NE + col] += v + bias[col];
      } else if constexpr (MODE == GM_GELU) {
        float x = v + bias[col];
        float gl = 0.5f * x * (1.0f + erff(x * 0.70710678118654752f));
        outH[(size_t)row * (4 * NE) + col] = (h16)gl;
      } else {  // GM_OUT
        outF[(size_t)row * NV + col] = v + bias[col];
      }
    }
  }
}

// ---------------------------------------------------------------- attention
// grid = (B*H, T/128), 8 waves/block; wave owns 16 query rows; flash softmax.
// Scores go C-layout -> LDS (A-layout); each lane then owns one full row
// half (16 K slots): serial max/sum + one shfl_xor(16) replaces 64 bpermutes.
__global__ __launch_bounds__(256) void attn_k(const h16* __restrict__ q,
                                              const h16* __restrict__ k,
                                              const h16* __restrict__ vT,
                                              h16* __restrict__ ao) {
  __shared__ __align__(32) h16 sbuf[8][16 * 32];
  __shared__ float cbuf[8][16];
  int lane = threadIdx.x & 31;
  int wave = threadIdx.x >> 5;
  int m = lane & 15;
  int hb = lane >> 4;
  int bh = blockIdx.x;
  int qTile = blockIdx.y * 128 + wave * 16;
  const h16* qp = q + (size_t)bh * NT * NHS;     // [T,HS]
  const h16* kp = k + (size_t)bh * NT * NHS;     // [T,HS] (= Bt for Q*K^T)
  const h16* vp = vT + (size_t)bh * NHS * NT;    // [HS,T] (= Bt for P*V)
  h16* pw = &sbuf[wave][0];
  float* cw = &cbuf[wave][0];

  v16h qf = ld_a16(qp + (size_t)(qTile + m) * NHS + 16 * hb);
  float rmax = -1e30f, rsum = 0.0f;              // stats for row r = m
  v8f o0 = {}, o1 = {};
  v8f z = {};
  const float sc = 0.17677669529663687f;         // 1/sqrt(32)

  int sEnd = qTile + 16;
  for (int s0 = 0; s0 < sEnd; s0 += 32) {
    v16h kf0 = ld_b16(kp + (size_t)(s0 + m) * NHS + 16 * hb);
    v16h kf1 = ld_b16(kp + (size_t)(s0 + 16 + m) * NHS + 16 * hb);
    v8f sA = wmma32(qf, kf0, z);
    v8f sB = wmma32(qf, kf1, z);
    // mask + scale, park scores in LDS at their A-layout position
#pragma unroll
    for (int i = 0; i < 8; ++i) {
      int rrow = 8 * hb + i;                     // q index = qTile + rrow
      float sa = (float)sA[i] * sc;
      float sb = (float)sB[i] * sc;
      if (s0 + m > qTile + rrow)      sa = -30000.0f;   // causal mask
      if (s0 + 16 + m > qTile + rrow) sb = -30000.0f;
      pw[rrow * 32 + m] = (h16)sa;
      pw[rrow * 32 + 16 + m] = (h16)sb;
    }
    asm volatile("" ::: "memory");               // keep DS store->load order
    v16h sf = ld_a16(pw + m * 32 + 16 * hb);     // row m, 16 K slots
    float sv[16];
#pragma unroll
    for (int j = 0; j < 16; ++j) sv[j] = (float)sf[j];
    float mx = sv[0];
#pragma unroll
    for (int j = 1; j < 16; ++j) mx = fmaxf(mx, sv[j]);
    mx = fmaxf(mx, __shfl_xor(mx, 16, 32));      // combine the two half-rows
    float nm = fmaxf(rmax, mx);
    float corr = __expf(rmax - nm);
    rmax = nm;
    v16h pf;
    float ls = 0.0f;
#pragma unroll
    for (int j = 0; j < 16; ++j) {
      float p = __expf(sv[j] - nm);
      ls += p;
      pf[j] = (h16)p;                            // A-fragment of P, in place
    }
    ls += __shfl_xor(ls, 16, 32);
    rsum = rsum * corr + ls;
    // broadcast per-row rescale factor to C-layout lanes
    cw[m] = corr;                                // lanes m and m+16 agree
    asm volatile("" ::: "memory");
    float4 c0 = *(const float4*)&cw[8 * hb];
    float4 c1 = *(const float4*)&cw[8 * hb + 4];
    o0[0] *= c0.x; o0[1] *= c0.y; o0[2] *= c0.z; o0[3] *= c0.w;
    o0[4] *= c1.x; o0[5] *= c1.y; o0[6] *= c1.z; o0[7] *= c1.w;
    o1[0] *= c0.x; o1[1] *= c0.y; o1[2] *= c0.z; o1[3] *= c0.w;
    o1[4] *= c1.x; o1[5] *= c1.y; o1[6] *= c1.z; o1[7] *= c1.w;

    v16h v0 = ld_b16(vp + (size_t)m * NT + s0 + 16 * hb);          // d 0..15
    v16h v1 = ld_b16(vp + (size_t)(16 + m) * NT + s0 + 16 * hb);   // d 16..31
    o0 = wmma32(pf, v0, o0);
    o1 = wmma32(pf, v1, o1);
  }

  cw[m] = rsum;
  asm volatile("" ::: "memory");
  float4 s0v = *(const float4*)&cw[8 * hb];
  float4 s1v = *(const float4*)&cw[8 * hb + 4];
  float inv[8] = {1.0f / s0v.x, 1.0f / s0v.y, 1.0f / s0v.z, 1.0f / s0v.w,
                  1.0f / s1v.x, 1.0f / s1v.y, 1.0f / s1v.z, 1.0f / s1v.w};
  int b = bh >> 2, h = bh & 3;
#pragma unroll
  for (int i = 0; i < 8; ++i) {
    int row = qTile + 8 * hb + i;
    size_t off = ((size_t)(b * NT + row)) * NE + h * NHS;
    ao[off + m] = (h16)(o0[i] * inv[i]);
    ao[off + 16 + m] = (h16)(o1[i] * inv[i]);
  }
}

// ---------------------------------------------------------------- launch
// workspace layout (bytes)
#define OFF_X     ((size_t)0)                        // fp32 residual [M,E]
#define OFF_XH    (OFF_X    + (size_t)NM*NE*4)       // f16 ln out   [M,E]
#define OFF_Q     (OFF_XH   + (size_t)NM*NE*2)       // f16 [B,H,T,HS]
#define OFF_K     (OFF_Q    + (size_t)NB*NH*NT*NHS*2)
#define OFF_VT    (OFF_K    + (size_t)NB*NH*NT*NHS*2)
#define OFF_AO    (OFF_VT   + (size_t)NB*NH*NT*NHS*2)
#define OFF_H2    OFF_Q                              // f16 [M,4E] overlaps q/k/vT/ao
#define OFF_WQKV  (OFF_AO   + (size_t)NM*NE*2)
#define OFF_WPROJ (OFF_WQKV + (size_t)NL*384*NE*2)
#define OFF_W1    (OFF_WPROJ+ (size_t)NL*NE*NE*2)
#define OFF_W2    (OFF_W1   + (size_t)NL*4*NE*NE*2)
#define OFF_WHEAD (OFF_W2   + (size_t)NL*4*NE*NE*2)

extern "C" void kernel_launch(void* const* d_in, const int* in_sizes, int n_in,
                              void* d_out, int out_size, void* d_ws,
                              size_t ws_size, hipStream_t stream) {
  (void)in_sizes; (void)n_in; (void)out_size; (void)ws_size;
  const int*   idx     = (const int*)  d_in[0];
  const float* tok_emb = (const float*)d_in[1];
  const float* pos_emb = (const float*)d_in[2];
  const float* wq      = (const float*)d_in[3];
  const float* wk      = (const float*)d_in[4];
  const float* wv      = (const float*)d_in[5];
  const float* w_proj  = (const float*)d_in[6];
  const float* b_proj  = (const float*)d_in[7];
  const float* ln1_g   = (const float*)d_in[8];
  const float* ln1_b   = (const float*)d_in[9];
  const float* ln2_g   = (const float*)d_in[10];
  const float* ln2_b   = (const float*)d_in[11];
  const float* w1      = (const float*)d_in[12];
  const float* b1      = (const float*)d_in[13];
  const float* w2      = (const float*)d_in[14];
  const float* b2      = (const float*)d_in[15];
  const float* lnf_g   = (const float*)d_in[16];
  const float* lnf_b   = (const float*)d_in[17];
  const float* w_head  = (const float*)d_in[18];
  const float* b_head  = (const float*)d_in[19];

  char* ws = (char*)d_ws;
  float* xf   = (float*)(ws + OFF_X);
  h16* xh     = (h16*)(ws + OFF_XH);
  h16* qb     = (h16*)(ws + OFF_Q);
  h16* kb     = (h16*)(ws + OFF_K);
  h16* vtb    = (h16*)(ws + OFF_VT);
  h16* aob    = (h16*)(ws + OFF_AO);
  h16* h2b    = (h16*)(ws + OFF_H2);
  h16* wqkvT  = (h16*)(ws + OFF_WQKV);
  h16* wprojh = (h16*)(ws + OFF_WPROJ);
  h16* w1h    = (h16*)(ws + OFF_W1);
  h16* w2h    = (h16*)(ws + OFF_W2);
  h16* wheadh = (h16*)(ws + OFF_WHEAD);

  // weight prep (runs every call; deterministic)
  pack_qkv_f16<<<(NL * 384 * NE + 255) / 256, 256, 0, stream>>>(wq, wk, wv, wqkvT);
  cast_f16<<<256, 256, 0, stream>>>(w_proj, wprojh, NL * NE * NE);
  cast_f16<<<1024, 256, 0, stream>>>(w1, w1h, NL * 4 * NE * NE);
  cast_f16<<<1024, 256, 0, stream>>>(w2, w2h, NL * 4 * NE * NE);
  cast_f16<<<256, 256, 0, stream>>>(w_head, wheadh, NV * NE);

  embed_k<<<(NM * (NE / 4)) / 256, 256, 0, stream>>>(idx, tok_emb, pos_emb, xf);

  for (int l = 0; l < NL; ++l) {
    ln_f16<<<NM / 8, 256, 0, stream>>>(xf, ln1_g + l * NE, ln1_b + l * NE, xh);
    gemm_ws<GM_QKV><<<dim3(NM / 128, 384 / 32), 256, 0, stream>>>(
        xh, wqkvT + (size_t)l * 384 * NE, nullptr, nullptr, nullptr,
        qb, kb, vtb, NE);
    attn_k<<<dim3(NB * NH, NT / 128), 256, 0, stream>>>(qb, kb, vtb, aob);
    gemm_ws<GM_RESID><<<dim3(NM / 128, NE / 32), 256, 0, stream>>>(
        aob, wprojh + (size_t)l * NE * NE, b_proj + l * NE, xf, nullptr,
        nullptr, nullptr, nullptr, NE);
    ln_f16<<<NM / 8, 256, 0, stream>>>(xf, ln2_g + l * NE, ln2_b + l * NE, xh);
    gemm_ws<GM_GELU><<<dim3(NM / 128, (4 * NE) / 32), 256, 0, stream>>>(
        xh, w1h + (size_t)l * 4 * NE * NE, b1 + l * 4 * NE, nullptr, h2b,
        nullptr, nullptr, nullptr, NE);
    gemm_ws<GM_RESID><<<dim3(NM / 128, NE / 32), 256, 0, stream>>>(
        h2b, w2h + (size_t)l * NE * 4 * NE, b2 + l * NE, xf, nullptr,
        nullptr, nullptr, nullptr, 4 * NE);
  }
  ln_f16<<<NM / 8, 256, 0, stream>>>(xf, lnf_g, lnf_b, xh);
  gemm_ws<GM_OUT><<<dim3(NM / 128, NV / 32), 256, 0, stream>>>(
      xh, wheadh, b_head, (float*)d_out, nullptr, nullptr, nullptr, nullptr, NE);
}
